// BinaryLabelSoftRouter_2104533975597
// MI455X (gfx1250) — matrix operand order repaired and checked by reading, not compiled, not run on hardware
//
#include <hip/hip_runtime.h>
#include <hip/hip_bf16.h>
#include <math.h>

// ---------------------------------------------------------------------------
// BinaryLabelSoftRouter for MI455X (gfx1250, wave32, WMMA)
// GEMM1: X[32768,2048] @ W1[2048,256]  -> LN -> GELU  (v_wmma_f32_16x16x32_f16)
//        double-buffered LDS; W1T tiles via global_load_async_to_lds_b128
// GEMM2: H1[32768,256] @ W2[256,128]   -> LN -> GELU -> head(128->2) -> softmax
// EMA  : segmented scan over T=2048 (0.9^256 ~ 2e-12 => segment carries exact)
// ---------------------------------------------------------------------------

typedef _Float16 h16x16 __attribute__((ext_vector_type(16)));
typedef _Float16 h16x8  __attribute__((ext_vector_type(8)));
typedef _Float16 h16x2  __attribute__((ext_vector_type(2)));
typedef float    f32x8  __attribute__((ext_vector_type(8)));

#define NTOK  32768
#define DDIM  2048
#define HD1   256
#define HD2   128

// ---- CDNA5 async copy: global -> LDS, no VGPR staging, tracked by ASYNCcnt
__device__ __forceinline__ void async_copy_b128(void* lds_dst, const void* gsrc) {
  uint32_t loff = (uint32_t)(uintptr_t)lds_dst;     // LDS aperture: low 32 bits = LDS offset
  uint64_t gaddr = (uint64_t)(uintptr_t)gsrc;
  asm volatile("global_load_async_to_lds_b128 %0, %1, off"
               :: "v"(loff), "v"(gaddr) : "memory");
}
__device__ __forceinline__ void async_copy_b32(void* lds_dst, const void* gsrc) {
  uint32_t loff = (uint32_t)(uintptr_t)lds_dst;
  uint64_t gaddr = (uint64_t)(uintptr_t)gsrc;
  asm volatile("global_load_async_to_lds_b32 %0, %1, off"
               :: "v"(loff), "v"(gaddr) : "memory");
}
__device__ __forceinline__ void wait_async0() {
  asm volatile("s_wait_asynccnt 0x0" ::: "memory");
}

// -------------------------- weight prep (fp32 -> f16, transposed) -----------
__global__ __launch_bounds__(256)
void prep_weights(const float* __restrict__ W1, const float* __restrict__ W2,
                  _Float16* __restrict__ w1t, _Float16* __restrict__ w2t) {
  int i = blockIdx.x * 256 + threadIdx.x;
  if (i < HD1 * DDIM) {                    // W1T[n][k] = W1[k][n]
    int n = i / DDIM, k = i % DDIM;
    w1t[i] = (_Float16)W1[k * HD1 + n];
  }
  if (i < HD2 * HD1) {                     // W2T[n][k] = W2[k][n]
    int n = i / HD1, k = i % HD1;
    w2t[i] = (_Float16)W2[k * HD2 + n];
  }
}

__device__ __forceinline__ float gelu_exact(float x) {
  return 0.5f * x * (1.0f + erff(x * 0.70710678118654752f));
}

// -------------------------- GEMM1 + LN + GELU -------------------------------
__global__ __launch_bounds__(256)
void gemm1_ln_gelu(const float* __restrict__ X, const _Float16* __restrict__ W1T,
                   const float* __restrict__ b1, const float* __restrict__ gam,
                   const float* __restrict__ bet, _Float16* __restrict__ H1out) {
  __shared__ __align__(16) _Float16 As[2][16 * 32];    // 16 rows x K32, row-major
  __shared__ __align__(16) _Float16 Bs[2][256 * 32];   // 256 cols x K32, N-major
  __shared__ __align__(16) float    Cs[16 * 256];

  const int tid  = threadIdx.x;
  const int lane = tid & 31;
  const int w    = tid >> 5;           // wave 0..7
  const int m0   = blockIdx.x * 16;
  const int g    = lane >> 4;          // lane half (ISA fragment layout)
  const int mr   = lane & 15;

  // X staging slot for this thread (2 fp32 -> 2 f16)
  const int xr = tid >> 4, xc = (tid & 15) * 2;
  const float* xsrc = X + (size_t)(m0 + xr) * DDIM + xc;

  // issue one k-tile of W1T (256x32 f16 = 1024 x 16B chunks) straight into LDS
  auto issueB = [&](int k0, int buf) {
    #pragma unroll
    for (int i = 0; i < 4; ++i) {
      int id = tid + 256 * i;
      int n = id >> 2, j = id & 3;
      async_copy_b128(&Bs[buf][n * 32 + j * 8],
                      W1T + (size_t)n * DDIM + k0 + j * 8);
    }
  };

  f32x8 acc0 = {};
  f32x8 acc1 = {};

  // ---- prologue: tile 0 into buffer 0
  issueB(0, 0);
  {
    float2 v = *(const float2*)xsrc;
    h16x2 hv; hv.x = (_Float16)v.x; hv.y = (_Float16)v.y;
    *(h16x2*)(&As[0][xr * 32 + xc]) = hv;
  }
  wait_async0();
  __syncthreads();

  for (int kt = 0; kt < DDIM / 32; ++kt) {
    const int cur = kt & 1, nxt = cur ^ 1;
    const bool more = (kt + 1) < (DDIM / 32);

    float2 xv;
    if (more) {                       // launch tile kt+1 while computing kt
      issueB((kt + 1) * 32, nxt);
      xv = *(const float2*)(xsrc + (size_t)(kt + 1) * 32);
      __builtin_prefetch(xsrc + (size_t)(kt + 2) * 32, 0, 3);
    }

    // A frag (16-bit A 16x32): lane half g holds K = [8g..8g+7] U [16+8g..23+8g]
    union { h16x16 v; h16x8 h[2]; } af, bf0, bf1;
    af.h[0] = *(const h16x8*)(&As[cur][mr * 32 + g * 8]);
    af.h[1] = *(const h16x8*)(&As[cur][mr * 32 + 16 + g * 8]);
    // B frag: lane half g holds K = [8g..8g+7] U [16+8g..23+8g] for its column
    const int nb0 = w * 32 + mr;
    const int nb1 = nb0 + 16;
    bf0.h[0] = *(const h16x8*)(&Bs[cur][nb0 * 32 + g * 8]);
    bf0.h[1] = *(const h16x8*)(&Bs[cur][nb0 * 32 + 16 + g * 8]);
    bf1.h[0] = *(const h16x8*)(&Bs[cur][nb1 * 32 + g * 8]);
    bf1.h[1] = *(const h16x8*)(&Bs[cur][nb1 * 32 + 16 + g * 8]);

    acc0 = __builtin_amdgcn_wmma_f32_16x16x32_f16(false, af.v, false, bf0.v,
                                                  (short)0, acc0, false, false);
    acc1 = __builtin_amdgcn_wmma_f32_16x16x32_f16(false, af.v, false, bf1.v,
                                                  (short)0, acc1, false, false);

    if (more) {                       // stage converted X tile into the nxt buffer
      h16x2 hv; hv.x = (_Float16)xv.x; hv.y = (_Float16)xv.y;
      *(h16x2*)(&As[nxt][xr * 32 + xc]) = hv;
    }
    wait_async0();
    __syncthreads();
  }

  {   // C layout: VGPR r, lane -> row = r + 8*(lane>>4), col = lane&15
    int c0 = w * 32 + mr, c1 = c0 + 16;
    float bb0 = b1[c0], bb1 = b1[c1];
    #pragma unroll
    for (int r = 0; r < 8; ++r) {
      Cs[(8 * g + r) * 256 + c0] = acc0[r] + bb0;
      Cs[(8 * g + r) * 256 + c1] = acc1[r] + bb1;
    }
  }
  __syncthreads();
  // LayerNorm + GELU: wave handles rows 2w, 2w+1; lane owns 8 columns
  for (int rr = 0; rr < 2; ++rr) {
    int row = w * 2 + rr;
    float vals[8], s = 0.f, ss = 0.f;
    #pragma unroll
    for (int j = 0; j < 8; ++j) {
      float v = Cs[row * 256 + lane * 8 + j];
      vals[j] = v; s += v; ss += v * v;
    }
    #pragma unroll
    for (int off = 16; off >= 1; off >>= 1) {
      s  += __shfl_xor(s, off, 32);
      ss += __shfl_xor(ss, off, 32);
    }
    float mean = s * (1.0f / 256.0f);
    float var  = ss * (1.0f / 256.0f) - mean * mean;
    float inv  = rsqrtf(var + 1e-5f);
    _Float16 ov[8];
    #pragma unroll
    for (int j = 0; j < 8; ++j) {
      int c = lane * 8 + j;
      float xh = (vals[j] - mean) * inv * gam[c] + bet[c];
      ov[j] = (_Float16)gelu_exact(xh);
    }
    *(h16x8*)(H1out + (size_t)(m0 + row) * 256 + lane * 8) = *(h16x8*)ov;
  }
}

// -------------------------- GEMM2 + LN + GELU + head ------------------------
__global__ __launch_bounds__(256)
void gemm2_head(const _Float16* __restrict__ H1in, const _Float16* __restrict__ W2T,
                const float* __restrict__ b2, const float* __restrict__ gam,
                const float* __restrict__ bet, const float* __restrict__ W3,
                const float* __restrict__ b3, const int* __restrict__ labels,
                const float* __restrict__ temperature,
                float* __restrict__ out_base, float* __restrict__ out_adj) {
  __shared__ __align__(16) _Float16 As[2][16 * 32];
  __shared__ __align__(16) _Float16 Bs[2][128 * 32];
  __shared__ __align__(16) float    Cs[16 * 128];

  const int tid  = threadIdx.x;
  const int lane = tid & 31;
  const int w    = tid >> 5;
  const int m0   = blockIdx.x * 16;
  const int g    = lane >> 4;
  const int mr   = lane & 15;

  const int xr = tid >> 4, xc = (tid & 15) * 2;

  // one k-tile: A (16x32 f16, async b32/thread) + B (128x32 f16, async b128)
  auto issueTile = [&](int k0, int buf) {
    async_copy_b32(&As[buf][xr * 32 + xc],
                   H1in + (size_t)(m0 + xr) * 256 + k0 + xc);
    #pragma unroll
    for (int i = 0; i < 2; ++i) {
      int id = tid + 256 * i;
      int n = id >> 2, j = id & 3;
      async_copy_b128(&Bs[buf][n * 32 + j * 8],
                      W2T + (size_t)n * 256 + k0 + j * 8);
    }
  };

  f32x8 acc = {};
  issueTile(0, 0);
  wait_async0();
  __syncthreads();

  for (int kt = 0; kt < HD1 / 32; ++kt) {
    const int cur = kt & 1, nxt = cur ^ 1;
    if ((kt + 1) < (HD1 / 32)) issueTile((kt + 1) * 32, nxt);

    union { h16x16 v; h16x8 h[2]; } af, bf;
    af.h[0] = *(const h16x8*)(&As[cur][mr * 32 + g * 8]);
    af.h[1] = *(const h16x8*)(&As[cur][mr * 32 + 16 + g * 8]);
    const int nb = w * 16 + mr;
    bf.h[0] = *(const h16x8*)(&Bs[cur][nb * 32 + g * 8]);
    bf.h[1] = *(const h16x8*)(&Bs[cur][nb * 32 + 16 + g * 8]);
    acc = __builtin_amdgcn_wmma_f32_16x16x32_f16(false, af.v, false, bf.v,
                                                 (short)0, acc, false, false);
    wait_async0();
    __syncthreads();
  }

  {
    int c = w * 16 + mr;
    float bb = b2[c];
    #pragma unroll
    for (int r = 0; r < 8; ++r) Cs[(8 * g + r) * 128 + c] = acc[r] + bb;
  }
  __syncthreads();

  const float temp = fmaxf(temperature[0], 0.1f);
  for (int rr = 0; rr < 2; ++rr) {
    int row = w * 2 + rr;
    // --- LayerNorm(128) + GELU: lane owns 4 columns
    float vals[4], s = 0.f, ss = 0.f;
    #pragma unroll
    for (int j = 0; j < 4; ++j) {
      float v = Cs[row * 128 + lane * 4 + j];
      vals[j] = v; s += v; ss += v * v;
    }
    #pragma unroll
    for (int off = 16; off >= 1; off >>= 1) {
      s  += __shfl_xor(s, off, 32);
      ss += __shfl_xor(ss, off, 32);
    }
    float mean = s * (1.0f / 128.0f);
    float var  = ss * (1.0f / 128.0f) - mean * mean;
    float inv  = rsqrtf(var + 1e-5f);
    // --- head 128->2: lane partials over its 4 elements
    float p0 = 0.f, p1 = 0.f;
    #pragma unroll
    for (int j = 0; j < 4; ++j) {
      int c = lane * 4 + j;
      float h = gelu_exact((vals[j] - mean) * inv * gam[c] + bet[c]);
      p0 += h * W3[c * 2 + 0];
      p1 += h * W3[c * 2 + 1];
    }
    #pragma unroll
    for (int off = 16; off >= 1; off >>= 1) {
      p0 += __shfl_xor(p0, off, 32);
      p1 += __shfl_xor(p1, off, 32);
    }
    if (lane == 0) {
      int gm = m0 + row;
      float a0 = tanhf(p0 + b3[0]) * 0.1f;
      float a1 = tanhf(p1 + b3[1]) * 0.1f;
      int   lb = labels[gm];
      float base0 = lb ? 0.25f : 0.75f;
      float base1 = lb ? 0.75f : 0.25f;
      float z0 = (base0 + a0) / temp, z1 = (base1 + a1) / temp;
      float zm = fmaxf(z0, z1);
      float e0 = __expf(z0 - zm), e1 = __expf(z1 - zm);
      float di = 1.0f / (e0 + e1);
      out_base[gm * 2 + 0] = base0;   out_base[gm * 2 + 1] = base1;
      out_adj [gm * 2 + 0] = e0 * di; out_adj [gm * 2 + 1] = e1 * di;
    }
  }
}

// -------------------------- segmented EMA scan ------------------------------
// 32 chains (B=16 x 2 comps), 8 segments of 256 each. 0.9^256 ~ 1.93e-12.
__global__ __launch_bounds__(256)
void ema_scan(const float* __restrict__ adj, const float* __restrict__ prev,
              float* __restrict__ fin) {
  __shared__ float lf[256];
  int tid = threadIdx.x;
  int ch  = tid >> 3;          // chain 0..31
  int s   = tid & 7;           // segment 0..7
  int b   = ch >> 1, c = ch & 1;
  const float* src = adj + ((size_t)b * 2048 + s * 256) * 2 + c;

  float v = 0.f;               // pass 1: local scan w/ zero carry -> segment final
  #pragma unroll 8
  for (int i = 0; i < 256; ++i) v = 0.9f * v + 0.1f * src[i * 2];
  lf[tid] = v;
  __syncthreads();

  const float F = 1.932e-12f;  // 0.9^256
  float carry = prev[b * 2 + c];
  for (int j = 0; j < s; ++j) carry = lf[ch * 8 + j] + F * carry;

  float* dst = fin + ((size_t)b * 2048 + s * 256) * 2 + c;
  v = carry;                   // pass 2: exact scan with propagated carry
  #pragma unroll 8
  for (int i = 0; i < 256; ++i) {
    v = 0.9f * v + 0.1f * src[i * 2];
    dst[i * 2] = v;
  }
}

// ---------------------------------------------------------------------------
extern "C" void kernel_launch(void* const* d_in, const int* in_sizes, int n_in,
                              void* d_out, int out_size, void* d_ws, size_t ws_size,
                              hipStream_t stream) {
  const int*   labels = (const int*)  d_in[0];
  const float* X      = (const float*)d_in[1];
  const float* prev   = (const float*)d_in[2];
  const float* W1     = (const float*)d_in[3];
  const float* b1     = (const float*)d_in[4];
  const float* g1v    = (const float*)d_in[5];
  const float* bt1    = (const float*)d_in[6];
  const float* W2     = (const float*)d_in[7];
  const float* b2     = (const float*)d_in[8];
  const float* g2v    = (const float*)d_in[9];
  const float* bt2    = (const float*)d_in[10];
  const float* W3     = (const float*)d_in[11];
  const float* b3     = (const float*)d_in[12];
  const float* temp   = (const float*)d_in[13];

  float* out_final = (float*)d_out;            // (B,T,2)
  float* out_base  = out_final + NTOK * 2;     // (B,T,2)
  float* out_adj   = out_final + NTOK * 4;     // (B,T,2)

  _Float16* w1t = (_Float16*)d_ws;             // 256*2048 halves (1 MB)
  _Float16* w2t = w1t + (size_t)HD1 * DDIM;    // 128*256 halves
  _Float16* h1  = w2t + (size_t)HD2 * HD1;     // 32768*256 halves (16 MB)

  hipLaunchKernelGGL(prep_weights, dim3((HD1 * DDIM + 255) / 256), dim3(256), 0,
                     stream, W1, W2, w1t, w2t);
  hipLaunchKernelGGL(gemm1_ln_gelu, dim3(NTOK / 16), dim3(256), 0, stream,
                     X, w1t, b1, g1v, bt1, h1);
  hipLaunchKernelGGL(gemm2_head, dim3(NTOK / 16), dim3(256), 0, stream,
                     h1, w2t, b2, g2v, bt2, W3, b3, labels, temp,
                     out_base, out_adj);
  hipLaunchKernelGGL(ema_scan, dim3(1), dim3(256), 0, stream,
                     out_adj, prev, out_final);
}
